// MultiHeadAttention_65395172048953
// MI455X (gfx1250) — compile-verified, matrix operand order checked by reading, not run on hardware
//
#include <hip/hip_runtime.h>
#include <cstdint>

// ---------------------------------------------------------------------------
// MultiHeadAttention forward for MI455X (gfx1250, wave32, WMMA + TDM).
//   B=2, L=2048, D_MODEL=1024, H=16, D_K=64
// Pipeline:
//   1) cast fp32 -> bf16 for q,k,v and the 4 weight matrices
//   2) WMMA GEMM projections (TDM-fed, double-buffered LDS)
//   3) flash-style attention with WMMA (TDM-fed K / V^T tiles)
//   4) WMMA GEMM output projection -> fp32 d_out
// All matmuls: v_wmma_f32_16x16x32_bf16 (f32 accumulate).
// All LDS tile staging: tensor_load_to_lds (TDM) with per-wave TENSORcnt
// waits and double-buffered overlap of DMA with WMMA compute.
// ---------------------------------------------------------------------------

typedef unsigned short u16;
typedef __attribute__((ext_vector_type(16))) __bf16 v16bf;
typedef __attribute__((ext_vector_type(8)))  float  v8f;
typedef __attribute__((ext_vector_type(4)))  unsigned int u32x4;
typedef __attribute__((ext_vector_type(8)))  int   i32x8;
typedef __attribute__((ext_vector_type(4)))  int   i32x4;

union FragU {
    uint4 q[2];   // two 16-byte chunks per lane == 8 VGPRs
    v16bf v;
};

__device__ __forceinline__ u16 f2bf(float f) {
    union { float f; unsigned int u; } x;
    x.f = f;
    unsigned int u = x.u;
    u += 0x7fffu + ((u >> 16) & 1u);   // round-to-nearest-even
    return (u16)(u >> 16);
}

__device__ __forceinline__ v8f wmma_bf16(const v16bf& a, const v16bf& b, const v8f& c) {
    return __builtin_amdgcn_wmma_f32_16x16x32_bf16(
        /*neg_a=*/false, a, /*neg_b=*/false, b,
        /*c_mod=*/(short)0, c, /*reuse_a=*/false, /*reuse_b=*/false);
}

// ---------------------------------------------------------------------------
// TDM 2D tile load: global (height x width_el u16, row stride stride_el u16)
// -> LDS at lds_byte_addr with per-row padding (pad_enable) so the LDS pitch
// matches our fragment indexing. D# per CDNA5 ISA ch.8:
//   group0: count=1, lds_addr, 57b global_addr, type=2
//   group1: data_size=1(2B), pad_enable, pad_interval/amount, dims/strides
// This toolchain's builtin takes 6 args (g0, g1, g2, g3, g4, cpol); groups
// 2..4 are zero for 2D tiles.
// ---------------------------------------------------------------------------
__device__ __forceinline__ void tdm_load_2d(
    unsigned lds_byte_addr, const void* gptr,
    unsigned width_el, unsigned height, unsigned stride_el,
    unsigned pad_interval_code, unsigned pad_amount_code)
{
    unsigned long long ga = (unsigned long long)(uintptr_t)gptr;
    u32x4 g0;
    g0[0] = 1u;                                              // count=1, user desc
    g0[1] = lds_byte_addr;                                   // lds_addr
    g0[2] = (unsigned)(ga & 0xffffffffu);                    // global_addr[31:0]
    g0[3] = (unsigned)((ga >> 32) & 0x1ffffffu) | (2u << 30); // addr[56:32] | type=2
    i32x8 g1;
    g1[0] = (int)((1u << 16)                                 // data_size = 2 bytes
                | (1u << 20)                                 // pad_enable
                | (pad_interval_code << 22)
                | (pad_amount_code << 25));
    g1[1] = (int)((width_el & 0xffffu) << 16);               // tensor_dim0[15:0]
    g1[2] = (int)((width_el >> 16) | ((height & 0xffffu) << 16)); // d0 hi | d1 lo
    g1[3] = (int)((height >> 16) | ((width_el & 0xffffu) << 16)); // d1 hi | tile_dim0
    g1[4] = (int)(height & 0xffffu);                         // tile_dim1 (tile_dim2=0)
    g1[5] = (int)stride_el;                                  // tensor_dim0_stride[31:0]
    g1[6] = 0;                                               // stride0 hi | stride1 lo
    g1[7] = 0;
    i32x4 z4 = {0, 0, 0, 0};
    i32x8 z8 = {0, 0, 0, 0, 0, 0, 0, 0};
    __builtin_amdgcn_tensor_load_to_lds(g0, g1, z4, z4, z8, 0);
}

// ---------------------------------------------------------------------------
// fp32 -> bf16 cast (grid-stride)
// ---------------------------------------------------------------------------
__global__ void mha_cast_bf16(const float* __restrict__ in, u16* __restrict__ out, int n) {
    int i = blockIdx.x * blockDim.x + threadIdx.x;
    int stride = gridDim.x * blockDim.x;
    for (; i < n; i += stride) out[i] = f2bf(in[i]);
}

// ---------------------------------------------------------------------------
// GEMM: Y = X @ W^T + bias.  X: MxK bf16 row-major, W: NxK bf16 row-major.
// TDM stages A/B tiles (128x32, padded to 40-u16 pitch) into double-buffered
// LDS: wave0 DMAs A, wave1 DMAs B, each waits its own TENSORcnt, one barrier
// per K-step, next tile's DMA overlaps current tile's 8 WMMAs per wave.
// mode 0: bf16 out at (B,H,L,64); mode 1: bf16 at (B,H,64,L); mode 2: fp32 (M,N)
// ---------------------------------------------------------------------------
__global__ __launch_bounds__(256) void mha_gemm_wmma(
    const u16* __restrict__ A, const u16* __restrict__ Bw,
    const float* __restrict__ bias, void* __restrict__ outp,
    int M, int N, int K, int mode)
{
    // pitch 40 u16 = 16 DW data + 4 DW TDM pad (interval code 3 = 16DW, amount code 3 = 4DW)
    __shared__ u16 As[2][128 * 40];
    __shared__ u16 Bs[2][128 * 40];

    const int tid   = threadIdx.x;
    const int wave  = tid >> 5;
    const int lane  = tid & 31;
    const int lo    = lane & 15;
    const int hf    = lane >> 4;
    const int waveM = wave >> 1;   // 0..3
    const int waveN = wave & 1;    // 0..1
    const int m0 = blockIdx.y * 128;
    const int n0 = blockIdx.x * 128;

    v8f acc[2][4];
#pragma unroll
    for (int mf = 0; mf < 2; ++mf)
#pragma unroll
        for (int nf = 0; nf < 4; ++nf)
#pragma unroll
            for (int r = 0; r < 8; ++r) acc[mf][nf][r] = 0.0f;

    const int ksteps = K >> 5;

    // prologue: DMA tile 0 into buffer 0
    if (wave == 0)
        tdm_load_2d((unsigned)(uintptr_t)(const void*)&As[0][0],
                    A + (size_t)m0 * K, 32, 128, (unsigned)K, 3, 3);
    else if (wave == 1)
        tdm_load_2d((unsigned)(uintptr_t)(const void*)&Bs[0][0],
                    Bw + (size_t)n0 * K, 32, 128, (unsigned)K, 3, 3);

    for (int kb = 0; kb < ksteps; ++kb) {
        const int cur = kb & 1;
        if (wave < 2) __builtin_amdgcn_s_wait_tensorcnt(0);
        __syncthreads();   // publish TDM-written buffer to all waves

        // overlap: DMA next tile into the other buffer while we compute
        if (kb + 1 < ksteps) {
            const int k1 = (kb + 1) << 5;
            if (wave == 0)
                tdm_load_2d((unsigned)(uintptr_t)(const void*)&As[cur ^ 1][0],
                            A + (size_t)m0 * K + k1, 32, 128, (unsigned)K, 3, 3);
            else if (wave == 1)
                tdm_load_2d((unsigned)(uintptr_t)(const void*)&Bs[cur ^ 1][0],
                            Bw + (size_t)n0 * K + k1, 32, 128, (unsigned)K, 3, 3);
        }

        const u16* Ab = As[cur];
        const u16* Bb = Bs[cur];
        FragU a[2], b[4];
#pragma unroll
        for (int mf = 0; mf < 2; ++mf) {
            int base = (waveM * 32 + mf * 16 + lo) * 40;
            a[mf].q[0] = *(const uint4*)(&Ab[base + hf * 8]);
            a[mf].q[1] = *(const uint4*)(&Ab[base + 16 + hf * 8]);
        }
#pragma unroll
        for (int nf = 0; nf < 4; ++nf) {
            int base = (waveN * 64 + nf * 16 + lo) * 40;
            b[nf].q[0] = *(const uint4*)(&Bb[base + hf * 8]);
            b[nf].q[1] = *(const uint4*)(&Bb[base + 16 + hf * 8]);
        }
#pragma unroll
        for (int mf = 0; mf < 2; ++mf)
#pragma unroll
            for (int nf = 0; nf < 4; ++nf)
                acc[mf][nf] = wmma_bf16(a[mf].v, b[nf].v, acc[mf][nf]);
    }

    // epilogue: C layout = VGPR r -> row (r + 8*hf), lane lo -> col
#pragma unroll
    for (int mf = 0; mf < 2; ++mf) {
#pragma unroll
        for (int nf = 0; nf < 4; ++nf) {
            const int n = n0 + waveN * 64 + nf * 16 + lo;
            const float bv = bias[n];
#pragma unroll
            for (int r = 0; r < 8; ++r) {
                const int m = m0 + waveM * 32 + mf * 16 + r + 8 * hf;
                const float val = acc[mf][nf][r] + bv;
                if (mode == 2) {
                    ((float*)outp)[(size_t)m * N + n] = val;
                } else {
                    const int bb = m >> 11, l = m & 2047;   // L = 2048
                    const int hh = n >> 6,  d = n & 63;     // D_K = 64
                    size_t idx = (mode == 0)
                        ? ((((size_t)bb * 16 + hh) * 2048 + l) * 64 + d)
                        : ((((size_t)bb * 16 + hh) * 64 + d) * 2048 + l);
                    ((u16*)outp)[idx] = f2bf(val);
                }
            }
        }
    }
}

// ---------------------------------------------------------------------------
// Flash-style attention.  Grid (L/128, B*H), 256 threads = 8 waves.
// Each wave owns 16 query rows; Q fragments live in registers.  K and V^T
// chunks (64x64, padded to 72-u16 pitch) are TDM-DMAed into double-buffered
// LDS (wave0: K, wave1: V^T).  Per chunk per wave: 8 WMMAs for S = Q K^T,
// online softmax (width-16 shfl row reductions), wave-local P transpose
// through LDS (in-order LDS + explicit s_wait_dscnt), 8 WMMAs for O += P V.
// ---------------------------------------------------------------------------
__global__ __launch_bounds__(256) void mha_attn_wmma(
    const u16* __restrict__ Qp, const u16* __restrict__ Kp,
    const u16* __restrict__ Vt, const unsigned char* __restrict__ mask,
    u16* __restrict__ AO)
{
    // pitch 72 u16 = 32 DW data + 4 DW TDM pad (interval code 4 = 32DW, amount code 3 = 4DW)
    __shared__ u16 Ks[2][64 * 72];
    __shared__ u16 Vs[2][64 * 72];
    __shared__ u16 Ps[8][16 * 72];

    const int tid  = threadIdx.x;
    const int wave = tid >> 5;
    const int lane = tid & 31;
    const int lo   = lane & 15;
    const int hf   = lane >> 4;
    const int bh = blockIdx.y;
    const int b  = bh >> 4;
    const int h  = bh & 15;
    const int q0 = blockIdx.x * 128 + wave * 16;

    const u16* Qh = Qp + (size_t)bh * 2048 * 64;
    const u16* Kh = Kp + (size_t)bh * 2048 * 64;
    const u16* Vh = Vt + (size_t)bh * 64 * 2048;
    const unsigned char* mb = mask + (size_t)b * 2048;

    // Q A-fragments (16 rows x d_k=64 -> 2 k-steps), direct from global
    FragU aQ[2];
    {
        const u16* qr = Qh + (size_t)(q0 + lo) * 64;
#pragma unroll
        for (int ks = 0; ks < 2; ++ks) {
            aQ[ks].q[0] = *(const uint4*)(qr + ks * 32 + hf * 8);
            aQ[ks].q[1] = *(const uint4*)(qr + ks * 32 + 16 + hf * 8);
        }
    }

    v8f accO[4];
    float mrow[8], lrow[8];
#pragma unroll
    for (int df = 0; df < 4; ++df)
#pragma unroll
        for (int r = 0; r < 8; ++r) accO[df][r] = 0.0f;
#pragma unroll
    for (int r = 0; r < 8; ++r) { mrow[r] = -1.0e30f; lrow[r] = 0.0f; }

    // prologue: DMA key-chunk 0
    if (wave == 0)
        tdm_load_2d((unsigned)(uintptr_t)(const void*)&Ks[0][0],
                    Kh, 64, 64, 64, 4, 3);
    else if (wave == 1)
        tdm_load_2d((unsigned)(uintptr_t)(const void*)&Vs[0][0],
                    Vh, 64, 64, 2048, 4, 3);

    for (int kb = 0; kb < 32; ++kb) {
        const int cur = kb & 1;
        if (wave < 2) __builtin_amdgcn_s_wait_tensorcnt(0);
        __syncthreads();   // publish current K/V chunk

        // overlap: DMA next chunk into the other buffer
        if (kb + 1 < 32) {
            if (wave == 0)
                tdm_load_2d((unsigned)(uintptr_t)(const void*)&Ks[cur ^ 1][0],
                            Kh + (size_t)(kb + 1) * 64 * 64, 64, 64, 64, 4, 3);
            else if (wave == 1)
                tdm_load_2d((unsigned)(uintptr_t)(const void*)&Vs[cur ^ 1][0],
                            Vh + (size_t)(kb + 1) * 64, 64, 64, 2048, 4, 3);
        }

        const u16* Kb = Ks[cur];
        const u16* Vb = Vs[cur];

        // S = Q K^T (scaled), 4 key fragments of 16
        v8f sf[4];
#pragma unroll
        for (int nf = 0; nf < 4; ++nf) {
            v8f s;
#pragma unroll
            for (int r = 0; r < 8; ++r) s[r] = 0.0f;
#pragma unroll
            for (int ks = 0; ks < 2; ++ks) {
                FragU bk;
                const int kr = (nf * 16 + lo) * 72 + ks * 32 + hf * 8;
                bk.q[0] = *(const uint4*)(&Kb[kr]);
                bk.q[1] = *(const uint4*)(&Kb[kr + 16]);
                s = wmma_bf16(aQ[ks].v, bk.v, s);
            }
            const unsigned char mm = mb[kb * 64 + nf * 16 + lo];
#pragma unroll
            for (int r = 0; r < 8; ++r) {
                float sv = s[r] * 0.125f;            // 1/sqrt(64)
                if (mm) sv = -3.0e38f;
                s[r] = sv;
            }
            sf[nf] = s;
        }

        // online softmax: row = (r + 8*hf), cols live across each 16-lane group
        float mx[8];
#pragma unroll
        for (int r = 0; r < 8; ++r)
            mx[r] = fmaxf(fmaxf(sf[0][r], sf[1][r]), fmaxf(sf[2][r], sf[3][r]));
#pragma unroll
        for (int off = 8; off >= 1; off >>= 1)
#pragma unroll
            for (int r = 0; r < 8; ++r)
                mx[r] = fmaxf(mx[r], __shfl_xor(mx[r], off, 16));

        float alpha[8], psum[8];
#pragma unroll
        for (int r = 0; r < 8; ++r) {
            const float mn = fmaxf(mrow[r], mx[r]);
            alpha[r] = __expf(mrow[r] - mn);
            mrow[r]  = mn;
            psum[r]  = 0.0f;
        }
#pragma unroll
        for (int nf = 0; nf < 4; ++nf)
#pragma unroll
            for (int r = 0; r < 8; ++r) {
                const float p = __expf(sf[nf][r] - mrow[r]);
                psum[r] += p;
                Ps[wave][(r + 8 * hf) * 72 + nf * 16 + lo] = f2bf(p);
            }
#pragma unroll
        for (int off = 8; off >= 1; off >>= 1)
#pragma unroll
            for (int r = 0; r < 8; ++r)
                psum[r] += __shfl_xor(psum[r], off, 16);
#pragma unroll
        for (int r = 0; r < 8; ++r) lrow[r] = lrow[r] * alpha[r] + psum[r];
#pragma unroll
        for (int df = 0; df < 4; ++df)
#pragma unroll
            for (int r = 0; r < 8; ++r) accO[df][r] *= alpha[r];

        // wave-local LDS RAW: LDS is in-order per wave; make the dependency explicit
        asm volatile("s_wait_dscnt 0" ::: "memory");

        // read P back as A fragments (16 rows x 64 keys -> 2 k-steps)
        FragU aP[2];
#pragma unroll
        for (int ks = 0; ks < 2; ++ks) {
            const int pr = lo * 72 + ks * 32 + hf * 8;
            aP[ks].q[0] = *(const uint4*)(&Ps[wave][pr]);
            aP[ks].q[1] = *(const uint4*)(&Ps[wave][pr + 16]);
        }
        // O += P V : B fragment = rows of V^T (contiguous along keys)
#pragma unroll
        for (int df = 0; df < 4; ++df)
#pragma unroll
            for (int ks = 0; ks < 2; ++ks) {
                FragU bv;
                const int vr = (df * 16 + lo) * 72 + ks * 32 + hf * 8;
                bv.q[0] = *(const uint4*)(&Vb[vr]);
                bv.q[1] = *(const uint4*)(&Vb[vr + 16]);
                accO[df] = wmma_bf16(aP[ks].v, bv.v, accO[df]);
            }
    }

    // finalize: O /= l, store bf16 at (B, L, H*64 + d)
#pragma unroll
    for (int r = 0; r < 8; ++r) lrow[r] = 1.0f / lrow[r];
#pragma unroll
    for (int df = 0; df < 4; ++df) {
        const int d = h * 64 + df * 16 + lo;
#pragma unroll
        for (int r = 0; r < 8; ++r) {
            const int qr = q0 + r + 8 * hf;
            AO[((size_t)b * 2048 + qr) * 1024 + d] = f2bf(accO[df][r] * lrow[r]);
        }
    }
}

// ---------------------------------------------------------------------------
// launcher
// ---------------------------------------------------------------------------
extern "C" void kernel_launch(void* const* d_in, const int* in_sizes, int n_in,
                              void* d_out, int out_size, void* d_ws, size_t ws_size,
                              hipStream_t stream) {
    (void)in_sizes; (void)n_in; (void)out_size; (void)ws_size;
    const float* q  = (const float*)d_in[0];
    const float* k  = (const float*)d_in[1];
    const float* v  = (const float*)d_in[2];
    const unsigned char* mask = (const unsigned char*)d_in[3];
    const float* Wq = (const float*)d_in[4];
    const float* bq = (const float*)d_in[5];
    const float* Wk = (const float*)d_in[6];
    const float* bk = (const float*)d_in[7];
    const float* Wv = (const float*)d_in[8];
    const float* bv = (const float*)d_in[9];
    const float* Wo = (const float*)d_in[10];
    const float* bo = (const float*)d_in[11];

    char* ws = (char*)d_ws;                       // 64 MB used
    u16* qb  = (u16*)(ws + (size_t)0);            // 8 MB each for q/k/v bf16
    u16* kb_ = (u16*)(ws + ((size_t)8  << 20));
    u16* vb  = (u16*)(ws + ((size_t)16 << 20));
    u16* wqb = (u16*)(ws + ((size_t)24 << 20));   // 2 MB each weight
    u16* wkb = (u16*)(ws + ((size_t)26 << 20));
    u16* wvb = (u16*)(ws + ((size_t)28 << 20));
    u16* wob = (u16*)(ws + ((size_t)30 << 20));
    u16* Qp  = (u16*)(ws + ((size_t)32 << 20));   // 8 MB each
    u16* Kp  = (u16*)(ws + ((size_t)40 << 20));
    u16* Vt  = (u16*)(ws + ((size_t)48 << 20));
    u16* AO  = (u16*)(ws + ((size_t)56 << 20));

    const int NQKV = 2 * 2048 * 1024;   // 4M elements
    const int NW   = 1024 * 1024;

    mha_cast_bf16<<<8192, 256, 0, stream>>>(q,  qb,  NQKV);
    mha_cast_bf16<<<8192, 256, 0, stream>>>(k,  kb_, NQKV);
    mha_cast_bf16<<<8192, 256, 0, stream>>>(v,  vb,  NQKV);
    mha_cast_bf16<<<2048, 256, 0, stream>>>(Wq, wqb, NW);
    mha_cast_bf16<<<2048, 256, 0, stream>>>(Wk, wkb, NW);
    mha_cast_bf16<<<2048, 256, 0, stream>>>(Wv, wvb, NW);
    mha_cast_bf16<<<2048, 256, 0, stream>>>(Wo, wob, NW);

    const dim3 gg(8, 32), bb(256);
    mha_gemm_wmma<<<gg, bb, 0, stream>>>(qb,  wqb, bq, Qp, 4096, 1024, 1024, 0);
    mha_gemm_wmma<<<gg, bb, 0, stream>>>(kb_, wkb, bk, Kp, 4096, 1024, 1024, 0);
    mha_gemm_wmma<<<gg, bb, 0, stream>>>(vb,  wvb, bv, Vt, 4096, 1024, 1024, 1);

    mha_attn_wmma<<<dim3(16, 32), 256, 0, stream>>>(Qp, Kp, Vt, mask, AO);

    mha_gemm_wmma<<<gg, bb, 0, stream>>>(AO, wob, bo, d_out, 4096, 1024, 1024, 2);
}
